// BatchedRuleExperts_7232724926955
// MI455X (gfx1250) — matrix operand order repaired and checked by reading, not compiled, not run on hardware
//
#include <hip/hip_runtime.h>
#include <hip/hip_bf16.h>
#include <math.h>

// ---------------------------------------------------------------------------
// BatchedRuleExperts for MI455X (gfx1250, wave32, WMMA).
//
// Tokens are grouped by rule so each expert's 256KB of f32 weights is read
// from HBM exactly once (256MB total -> ~11us at 23.3 TB/s); the two GEMMs
// per rule run on v_wmma_f32_16x16x32_bf16 over 16-token tiles, with weights
// staged transposed in LDS as bf16.
// ---------------------------------------------------------------------------

#define NUM_RULES   1024
#define D_MODEL     128
#define EXPERT_DIM  256

// LDS strides (in ushort elements), padded by 8 (16B) to spread banks.
#define W1T_S (D_MODEL + 8)      // w1t[e][d]   : 256 rows
#define W2T_S (EXPERT_DIM + 8)   // w2t[d][e]   : 128 rows
#define XT_S  (D_MODEL + 8)      // xt[m][d]    : 16 rows
#define HT_S  (EXPERT_DIM + 8)   // ht[m][e]    : 16 rows

typedef __attribute__((ext_vector_type(16))) __bf16 v16bf;
typedef __attribute__((ext_vector_type(8)))  float  v8f;

union Frag {           // 32 bytes: one WMMA A/B fragment per lane
    uint4 q[2];
    v16bf v;
};

// float -> bf16 bits with round-to-nearest-even
__device__ __forceinline__ unsigned short f2us(float f) {
    union { float f; unsigned u; } a; a.f = f;
    unsigned r = a.u + 0x7FFFu + ((a.u >> 16) & 1u);
    return (unsigned short)(r >> 16);
}

// pack two floats into a bf16x2 dword
__device__ __forceinline__ unsigned pack2(float lo, float hi) {
    return (unsigned)f2us(lo) | ((unsigned)f2us(hi) << 16);
}

// Branch-free exact-ish GELU: 0.5*x*(1+erf(x/sqrt(2))).
// erf via Abramowitz-Stegun 7.1.26 (|err| < 1.5e-7), no divergence:
// one v_rcp, one v_exp, straight-line FMAs, sign via copysign.
__device__ __forceinline__ float gelu_erf(float v) {
    float z  = 0.70710678118654752f * v;
    float az = __builtin_fabsf(z);
    float t  = __builtin_amdgcn_rcpf(1.0f + 0.3275911f * az);
    float p  = 1.061405429f;
    p = p * t - 1.453152027f;
    p = p * t + 1.421413741f;
    p = p * t - 0.284496736f;
    p = p * t + 0.254829592f;
    p = p * t;
    float e    = __builtin_amdgcn_exp2f(az * az * -1.4426950408889634f);
    float erfa = 1.0f - p * e;                    // erf(|z|)
    float erfz = __builtin_copysignf(erfa, z);
    return 0.5f * v * (1.0f + erfz);
}

// ---------------------------- bucketing kernels ----------------------------

__global__ void init_ws_kernel(int* __restrict__ ws, int n) {
    int i = blockIdx.x * blockDim.x + threadIdx.x;
    if (i < n) ws[i] = 0;
}

__global__ void count_kernel(const int* __restrict__ rules,
                             int* __restrict__ counts, int n) {
    int i = blockIdx.x * blockDim.x + threadIdx.x;
    if (i < n) atomicAdd(&counts[rules[i]], 1);
}

__global__ __launch_bounds__(NUM_RULES)
void scan_kernel(const int* __restrict__ counts, int* __restrict__ offsets) {
    __shared__ int s[NUM_RULES];
    int t = threadIdx.x;
    s[t] = counts[t];
    __syncthreads();
    for (int o = 1; o < NUM_RULES; o <<= 1) {
        int v = (t >= o) ? s[t - o] : 0;
        __syncthreads();
        s[t] += v;
        __syncthreads();
    }
    offsets[t] = s[t] - counts[t];   // exclusive prefix
}

__global__ void scatter_kernel(const int* __restrict__ rules,
                               const int* __restrict__ offsets,
                               int* __restrict__ cursors,
                               int* __restrict__ perm, int n) {
    int i = blockIdx.x * blockDim.x + threadIdx.x;
    if (i < n) {
        int r = rules[i];
        int p = atomicAdd(&cursors[r], 1);
        perm[offsets[r] + p] = i;
    }
}

// ------------------------------ FFN kernel ---------------------------------
// One block per rule. 256 threads = 8 waves. Weights staged once as bf16 in
// LDS (transposed so each lane's WMMA B fragment is two 8-element K runs).

__global__ __launch_bounds__(256)
void ffn_kernel(const float* __restrict__ x,
                const float* __restrict__ w1, const float* __restrict__ b1,
                const float* __restrict__ w2, const float* __restrict__ b2,
                const int* __restrict__ counts, const int* __restrict__ offsets,
                const int* __restrict__ perm, float* __restrict__ out) {
    const int r     = blockIdx.x;
    const int count = counts[r];
    if (count == 0) return;
    const int off   = offsets[r];

    extern __shared__ char smem[];
    unsigned short* w1t = (unsigned short*)smem;            // [256][W1T_S]
    unsigned short* w2t = w1t + EXPERT_DIM * W1T_S;         // [128][W2T_S]
    unsigned short* xt  = w2t + D_MODEL   * W2T_S;          // [16][XT_S]
    unsigned short* ht  = xt  + 16        * XT_S;           // [16][HT_S]
    int*            toks = (int*)(ht + 16 * HT_S);          // [16]

    const int tid  = threadIdx.x;
    const int lane = tid & 31;
    const int wave = tid >> 5;
    const int half = lane >> 4;    // which 16-lane half of the wave
    const int ln16 = lane & 15;

    // ---- stage W1 (memory: [d][e], e contiguous) -> w1t[e][d] as bf16 ----
    // Each thread reads the same 4 e's from two adjacent d rows (both
    // coalesced float4 loads) and emits packed bf16x2 b32 stores.
    {
        const float4* w1v = (const float4*)(w1 + (size_t)r * D_MODEL * EXPERT_DIM);
        #pragma unroll 4
        for (int i = tid; i < (D_MODEL / 2) * (EXPERT_DIM / 4); i += 256) {
            int d0 = (i >> 6) << 1;          // 0,2,...,126
            int e0 = (i & 63) << 2;          // 0,4,...,252
            float4 a = w1v[(d0 * EXPERT_DIM + e0) >> 2];
            float4 b = w1v[((d0 + 1) * EXPERT_DIM + e0) >> 2];
            *(unsigned*)(w1t + (e0 + 0) * W1T_S + d0) = pack2(a.x, b.x);
            *(unsigned*)(w1t + (e0 + 1) * W1T_S + d0) = pack2(a.y, b.y);
            *(unsigned*)(w1t + (e0 + 2) * W1T_S + d0) = pack2(a.z, b.z);
            *(unsigned*)(w1t + (e0 + 3) * W1T_S + d0) = pack2(a.w, b.w);
        }
    }
    // ---- stage W2 (memory: [e][d], d contiguous) -> w2t[d][e] as bf16 ----
    {
        const float4* w2v = (const float4*)(w2 + (size_t)r * EXPERT_DIM * D_MODEL);
        #pragma unroll 4
        for (int i = tid; i < (EXPERT_DIM / 2) * (D_MODEL / 4); i += 256) {
            int e0 = (i >> 5) << 1;          // 0,2,...,254
            int d0 = (i & 31) << 2;          // 0,4,...,124
            float4 a = w2v[(e0 * D_MODEL + d0) >> 2];
            float4 b = w2v[((e0 + 1) * D_MODEL + d0) >> 2];
            *(unsigned*)(w2t + (d0 + 0) * W2T_S + e0) = pack2(a.x, b.x);
            *(unsigned*)(w2t + (d0 + 1) * W2T_S + e0) = pack2(a.y, b.y);
            *(unsigned*)(w2t + (d0 + 2) * W2T_S + e0) = pack2(a.z, b.z);
            *(unsigned*)(w2t + (d0 + 3) * W2T_S + e0) = pack2(a.w, b.w);
        }
    }

    // per-wave bias registers (wave owns e-tiles {wave, wave+8} and d-tile wave)
    const float b1a = b1[(size_t)r * EXPERT_DIM + (wave + 0) * 16 + ln16];
    const float b1b = b1[(size_t)r * EXPERT_DIM + (wave + 8) * 16 + ln16];
    const float b2v = b2[(size_t)r * D_MODEL   +  wave      * 16 + ln16];

    __syncthreads();

    const int ntiles = (count + 15) >> 4;
    for (int tile = 0; tile < ntiles; ++tile) {
        // ---- stage x tile (16 x 128) as bf16 + publish token ids ----
        {
            int row = tid >> 4;            // 0..15
            int col = tid & 15;
            int gi  = tile * 16 + row;
            int tok = (gi < count) ? perm[off + gi] : -1;
            if (col == 0) toks[row] = tok;

            int cg = col * 8;              // 8 floats per thread
            uint4 pk = make_uint4(0u, 0u, 0u, 0u);
            if (tok >= 0) {
                const float4* xp = (const float4*)(x + (size_t)tok * D_MODEL + cg);
                float4 v0 = xp[0], v1 = xp[1];
                pk.x = pack2(v0.x, v0.y);
                pk.y = pack2(v0.z, v0.w);
                pk.z = pack2(v1.x, v1.y);
                pk.w = pack2(v1.z, v1.w);
            }
            *(uint4*)(xt + row * XT_S + cg) = pk;   // one ds_store_b128
        }
        __syncthreads();

        // ---- layer 1: h = gelu(x @ W1 + b1)   M=16 K=128 N=256 ----
        v16bf afrag[4];
        #pragma unroll
        for (int q = 0; q < 4; ++q) {
            const unsigned short* p = xt + ln16 * XT_S + q * 32 + half * 8;
            Frag f;
            f.q[0] = *(const uint4*)(p);        // K = q*32 + half*8 .. +7
            f.q[1] = *(const uint4*)(p + 16);   // K = q*32 + 16 + half*8 .. +7
            afrag[q] = f.v;
        }
        #pragma unroll
        for (int t = 0; t < 2; ++t) {
            const int te = wave + t * 8;        // e-tile index (0..15)
            v8f c = {};
            #pragma unroll
            for (int q = 0; q < 4; ++q) {
                const unsigned short* bp =
                    w1t + (te * 16 + ln16) * W1T_S + q * 32 + half * 8;
                Frag fb;
                fb.q[0] = *(const uint4*)(bp);      // K = q*32 + half*8 .. +7
                fb.q[1] = *(const uint4*)(bp + 16); // K = q*32 + 16 + half*8 .. +7
                c = __builtin_amdgcn_wmma_f32_16x16x32_bf16(
                        false, afrag[q], false, fb.v, (short)0, c, false, false);
            }
            const float bb = (t == 0) ? b1a : b1b;
            #pragma unroll
            for (int v = 0; v < 8; ++v) {
                float val = c[v] + bb;
                ht[(v + half * 8) * HT_S + te * 16 + ln16] = f2us(gelu_erf(val));
            }
        }
        __syncthreads();

        // ---- layer 2: out = h @ W2 + b2   M=16 K=256 N=128 ----
        {
            const int td = wave;                // d-tile index (0..7)
            v8f c = {};
            #pragma unroll
            for (int q = 0; q < 8; ++q) {
                const unsigned short* ap = ht + ln16 * HT_S + q * 32 + half * 8;
                Frag fa;
                fa.q[0] = *(const uint4*)(ap);
                fa.q[1] = *(const uint4*)(ap + 16);
                const unsigned short* bp =
                    w2t + (td * 16 + ln16) * W2T_S + q * 32 + half * 8;
                Frag fb;
                fb.q[0] = *(const uint4*)(bp);
                fb.q[1] = *(const uint4*)(bp + 16);
                c = __builtin_amdgcn_wmma_f32_16x16x32_bf16(
                        false, fa.v, false, fb.v, (short)0, c, false, false);
            }
            #pragma unroll
            for (int v = 0; v < 8; ++v) {
                int tok = toks[v + half * 8];
                if (tok >= 0)
                    out[(size_t)tok * D_MODEL + td * 16 + ln16] = c[v] + b2v;
            }
        }
        __syncthreads();   // protect toks/xt/ht before next tile
    }
}

// ------------------------------- launcher ----------------------------------

extern "C" void kernel_launch(void* const* d_in, const int* in_sizes, int n_in,
                              void* d_out, int out_size, void* d_ws, size_t ws_size,
                              hipStream_t stream) {
    const float* x     = (const float*)d_in[0];
    const int*   rules = (const int*)  d_in[1];
    const float* w1    = (const float*)d_in[2];
    const float* b1    = (const float*)d_in[3];
    const float* w2    = (const float*)d_in[4];
    const float* b2    = (const float*)d_in[5];
    float*       out   = (float*)d_out;

    const int N = in_sizes[1];        // number of tokens (8192)

    int* ws      = (int*)d_ws;
    int* counts  = ws;                // [1024]
    int* offsets = ws + NUM_RULES;    // [1024]
    int* cursors = ws + 2 * NUM_RULES;// [1024]
    int* perm    = ws + 3 * NUM_RULES;// [N]

    init_ws_kernel<<<(3 * NUM_RULES + 255) / 256, 256, 0, stream>>>(ws, 3 * NUM_RULES);
    count_kernel  <<<(N + 255) / 256, 256, 0, stream>>>(rules, counts, N);
    scan_kernel   <<<1, NUM_RULES, 0, stream>>>(counts, offsets);
    scatter_kernel<<<(N + 255) / 256, 256, 0, stream>>>(rules, offsets, cursors, perm, N);

    const size_t smem =
        (size_t)(EXPERT_DIM * W1T_S + D_MODEL * W2T_S + 16 * XT_S + 16 * HT_S)
            * sizeof(unsigned short)
        + 16 * sizeof(int);           // ~150 KB -> 2 workgroups per WGP

    ffn_kernel<<<NUM_RULES, 256, smem, stream>>>(x, w1, b1, w2, b2,
                                                 counts, offsets, perm, out);
}
